// Mnn_Conv2d_Compose_without_Rho_32916629356637
// MI455X (gfx1250) — compile-verified, matrix operand order checked by reading, not run on hardware
//
#include <hip/hip_runtime.h>
#include <hip/hip_bf16.h>
#include <math.h>

typedef __attribute__((ext_vector_type(16))) _Float16 v16h;
typedef __attribute__((ext_vector_type(8)))  _Float16 v8h;
typedef __attribute__((ext_vector_type(8)))  float    v8f;

#define CIN   128
#define COUT  256
#define HW    56
#define IMG   3136      // 56*56
#define BATCH 32
#define MTOT  100352    // 32*3136
#define OUTHW 28
#define OUT1  6422528   // 32*256*28*28

// Async global->LDS copies (CDNA5, ASYNCcnt). Flip to 0 to fall back to
// synchronous uint4 copies through VGPRs if the assembler rejects them.
#define USE_ASYNC 1

// ---------------------------------------------------------------------------
// pack_x: NCHW f32 -> NHWC f16 for mean, and std^2 for the variance conv.
// LDS-tiled transpose: 128 channels x 64 spatial positions per block.
// ---------------------------------------------------------------------------
__global__ __launch_bounds__(256) void pack_x(
    const float* __restrict__ mean, const float* __restrict__ stdv,
    _Float16* __restrict__ Xm, _Float16* __restrict__ Xv)
{
    __shared__ float tile[128][65];
    const int tid = threadIdx.x;
    const int s0  = (blockIdx.x % 49) * 64;
    const int b   = blockIdx.x / 49;

    // ---- mean ----
    #pragma unroll 4
    for (int j = 0; j < 32; ++j) {
        int lin = tid + j * 256;
        int c = lin >> 6, s = lin & 63;
        tile[c][s] = mean[(size_t)(b * 128 + c) * IMG + s0 + s];
    }
    __syncthreads();
    #pragma unroll 4
    for (int j = 0; j < 32; ++j) {
        int lin = tid + j * 256;
        int s = lin >> 7, c = lin & 127;
        Xm[((size_t)b * IMG + s0 + s) * CIN + c] = (_Float16)tile[c][s];
    }
    __syncthreads();
    // ---- std^2 ----
    #pragma unroll 4
    for (int j = 0; j < 32; ++j) {
        int lin = tid + j * 256;
        int c = lin >> 6, s = lin & 63;
        float v = stdv[(size_t)(b * 128 + c) * IMG + s0 + s];
        tile[c][s] = v * v;
    }
    __syncthreads();
    #pragma unroll 4
    for (int j = 0; j < 32; ++j) {
        int lin = tid + j * 256;
        int s = lin >> 7, c = lin & 127;
        Xv[((size_t)b * IMG + s0 + s) * CIN + c] = (_Float16)tile[c][s];
    }
}

// ---------------------------------------------------------------------------
// pack_w: (Cout,Cin,3,3) f32 -> [tap][Cout][Cin] f16; w^2 pre-scaled by 4096.
// Also zero-fills the OOB "zero row" appended at X[MTOT] in both X buffers.
// ---------------------------------------------------------------------------
__global__ __launch_bounds__(256) void pack_w(
    const float* __restrict__ w, _Float16* __restrict__ Wm, _Float16* __restrict__ Wv,
    _Float16* __restrict__ Xm, _Float16* __restrict__ Xv)
{
    if (blockIdx.x == 0 && threadIdx.x < CIN) {
        Xm[(size_t)MTOT * CIN + threadIdx.x] = (_Float16)0.0f;
        Xv[(size_t)MTOT * CIN + threadIdx.x] = (_Float16)0.0f;
    }
    int o = blockIdx.x * 256 + threadIdx.x;
    if (o >= 9 * COUT * CIN) return;
    int cin = o & 127;
    int r   = o >> 7;
    int cout = r & 255;
    int tap  = r >> 8;
    float v = w[((size_t)(cout * CIN + cin)) * 9 + tap];
    Wm[o] = (_Float16)v;
    Wv[o] = (_Float16)(v * v * 4096.0f);
}

// ---------------------------------------------------------------------------
// WMMA fragment load from an LDS row (row-major [row][K], K padded 32->40 f16)
// 16-bit A-layout per ISA: element i -> K = 8*lh + i (i<8), 16 + 8*lh + (i-8).
// Two 16-byte DS loads per fragment.
// ---------------------------------------------------------------------------
__device__ __forceinline__ v16h load_frag(const _Float16* row, int lh)
{
    v8h lo = *reinterpret_cast<const v8h*>(row + lh * 8);
    v8h hi = *reinterpret_cast<const v8h*>(row + 16 + lh * 8);
    return __builtin_shufflevector(lo, hi, 0,1,2,3,4,5,6,7,8,9,10,11,12,13,14,15);
}

// ---------------------------------------------------------------------------
// Implicit-GEMM conv: out[M=B*H*W][N=Cout] = im2col(X) * Wt, K = 9 taps x 128.
// Block = 256 threads = 8 waves; 128x128 tile; wave = 32x64 (2x4 fragments).
// Fully unrolled 36 K-stages, double-buffered LDS, async global->LDS copies
// overlapped with v_wmma_f32_16x16x32_f16.
// ---------------------------------------------------------------------------
__global__ __launch_bounds__(256) void gemm_conv(
    const _Float16* __restrict__ X,    // [B][56][56][128] f16 (+ zero row at MTOT)
    const _Float16* __restrict__ Wt,   // [9][256][128]    f16
    const float*    __restrict__ bias, // [256] or nullptr
    float*          __restrict__ out,  // [M][256] f32
    float alpha, int has_bias)
{
    __shared__ __align__(16) _Float16 shA[2][128 * 40];
    __shared__ __align__(16) _Float16 shB[2][128 * 40];

    const int tid   = threadIdx.x;
    const int m0    = blockIdx.x * 128;
    const int n0    = blockIdx.y * 128;
    const int wave  = tid >> 5;
    const int lane  = tid & 31;
    const int lh    = lane >> 4;
    const int ln    = lane & 15;
    const int waveM = wave & 3;   // 4 M-subtiles of 32 rows
    const int waveN = wave >> 2;  // 2 N-subtiles of 64 cols

    // tile-load geometry: thread covers rows r0 and r0+64, 16B segment `seg`
    const int seg    = tid & 3;
    const int r0     = tid >> 2;
    const int segoff = seg * 8;

    // Hoisted im2col geometry: per-tap A row index (or zero row) per i.
    int moff[9][2];
    #pragma unroll
    for (int i = 0; i < 2; ++i) {
        const int m   = m0 + r0 + 64 * i;
        const int rem = m % IMG;
        const int y   = rem / HW;
        const int x   = rem % HW;
        #pragma unroll
        for (int tap = 0; tap < 9; ++tap) {
            const int dy = tap / 3 - 1;
            const int dx = tap % 3 - 1;
            const int ys = y + dy, xs = x + dx;
            moff[tap][i] = ((unsigned)ys < (unsigned)HW && (unsigned)xs < (unsigned)HW)
                         ? (m + dy * HW + dx) : MTOT;   // MTOT -> zero row
        }
    }

    v8f acc[2][4];
    #pragma unroll
    for (int mf = 0; mf < 2; ++mf)
        #pragma unroll
        for (int nf = 0; nf < 4; ++nf)
            acc[mf][nf] = (v8f){0.f,0.f,0.f,0.f,0.f,0.f,0.f,0.f};

    // Issue global->LDS copies for stage s (tap = s>>2, kc = (s&3)*32).
    auto issue = [&](int s) {
        const int tap = s >> 2;
        const int kc  = (s & 3) * 32;
        const int bi  = s & 1;
        #pragma unroll
        for (int i = 0; i < 2; ++i) {
            const int r = r0 + 64 * i;
            const _Float16* ga = X + (size_t)moff[tap][i] * CIN + kc + segoff;
            const _Float16* gb = Wt + ((size_t)tap * COUT + n0 + r) * CIN + kc + segoff;
#if USE_ASYNC
            const unsigned la = (unsigned)(uintptr_t)&shA[bi][r * 40 + segoff];
            const unsigned lb = (unsigned)(uintptr_t)&shB[bi][r * 40 + segoff];
            asm volatile("global_load_async_to_lds_b128 %0, %1, off"
                         :: "v"(la), "v"(ga) : "memory");
            asm volatile("global_load_async_to_lds_b128 %0, %1, off"
                         :: "v"(lb), "v"(gb) : "memory");
#else
            *reinterpret_cast<uint4*>(&shA[bi][r * 40 + segoff]) =
                *reinterpret_cast<const uint4*>(ga);
            *reinterpret_cast<uint4*>(&shB[bi][r * 40 + segoff]) =
                *reinterpret_cast<const uint4*>(gb);
#endif
        }
    };

    issue(0);
#if USE_ASYNC
    asm volatile("s_wait_asynccnt 0x0" ::: "memory");
#endif
    __syncthreads();

    #pragma unroll
    for (int s = 0; s < 36; ++s) {
        if (s + 1 < 36) issue(s + 1);   // fill other buffer while we compute

        const _Float16* bufA = shA[s & 1];
        const _Float16* bufB = shB[s & 1];
        v16h afrag[2], bfrag[4];
        #pragma unroll
        for (int mf = 0; mf < 2; ++mf)
            afrag[mf] = load_frag(bufA + (waveM * 32 + mf * 16 + ln) * 40, lh);
        #pragma unroll
        for (int nf = 0; nf < 4; ++nf)
            bfrag[nf] = load_frag(bufB + (waveN * 64 + nf * 16 + ln) * 40, lh);

        #pragma unroll
        for (int mf = 0; mf < 2; ++mf)
            #pragma unroll
            for (int nf = 0; nf < 4; ++nf)
                acc[mf][nf] = __builtin_amdgcn_wmma_f32_16x16x32_f16(
                    false, afrag[mf], false, bfrag[nf],
                    (short)0, acc[mf][nf], false, false);

#if USE_ASYNC
        asm volatile("s_wait_asynccnt 0x0" ::: "memory");
#endif
        __syncthreads();
    }

    // epilogue: D layout -> VGPR r holds row (r + 8*lh), col = ln within frag
    #pragma unroll
    for (int mf = 0; mf < 2; ++mf) {
        #pragma unroll
        for (int nf = 0; nf < 4; ++nf) {
            const int gn = n0 + waveN * 64 + nf * 16 + ln;
            const float bv = has_bias ? bias[gn] : 0.0f;
            #pragma unroll
            for (int r = 0; r < 8; ++r) {
                const int gm = m0 + waveM * 32 + mf * 16 + lh * 8 + r;
                out[(size_t)gm * COUT + gn] = acc[mf][nf][r] * alpha + bv;
            }
        }
    }
}

// ---------------------------------------------------------------------------
// Deterministic per-channel BN stats: stage1 partials, stage2 finalize to
// affine constants a_scale/a_shift/s_scale.
// ---------------------------------------------------------------------------
__global__ __launch_bounds__(256) void reduce_stage1(
    const float* __restrict__ m_buf, float* __restrict__ partials)
{
    const int c = threadIdx.x;
    const size_t base = (size_t)blockIdx.x * 256 * COUT;
    float s = 0.f, q = 0.f;
    for (int r = 0; r < 256; ++r) {
        float v = m_buf[base + (size_t)r * COUT + c];
        s += v; q += v * v;
    }
    partials[(size_t)blockIdx.x * 512 + c]       = s;
    partials[(size_t)blockIdx.x * 512 + 256 + c] = q;
}

__global__ __launch_bounds__(256) void reduce_stage2(
    const float* __restrict__ partials,
    const float* __restrict__ gamma, const float* __restrict__ beta,
    float* __restrict__ a_scale, float* __restrict__ a_shift,
    float* __restrict__ s_scale)
{
    const int c = threadIdx.x;
    float s = 0.f, q = 0.f;
    for (int i = 0; i < 392; ++i) {
        s += partials[(size_t)i * 512 + c];
        q += partials[(size_t)i * 512 + 256 + c];
    }
    const float invM = 1.0f / (float)MTOT;
    float mu  = s * invM;
    float var = fmaxf(q * invM - mu * mu, 0.0f);
    float inv = rsqrtf(var + 1e-5f);
    float g   = gamma[c];
    a_scale[c] = inv * g;
    a_shift[c] = beta[c] - mu * inv * g;
    s_scale[c] = inv * fabsf(g);
}

// ---------------------------------------------------------------------------
// Fused erf-composition + 2x2 average pool; channel-last reads, NCHW writes.
// ---------------------------------------------------------------------------
__global__ __launch_bounds__(256) void compose_pool(
    const float* __restrict__ m_buf, const float* __restrict__ var_buf,
    const float* __restrict__ a_scale, const float* __restrict__ a_shift,
    const float* __restrict__ s_scale, float* __restrict__ outp)
{
    const int c   = threadIdx.x;
    int blk = blockIdx.x;             // 32*28*28
    const int px = blk % OUTHW; blk /= OUTHW;
    const int py = blk % OUTHW;
    const int b  = blk / OUTHW;

    const float asc = a_scale[c], ash = a_shift[c], ssc = s_scale[c];
    float su = 0.f, sq = 0.f;
    #pragma unroll
    for (int j = 0; j < 4; ++j) {
        const int y = 2 * py + (j >> 1);
        const int x = 2 * px + (j & 1);
        const size_t idx = ((size_t)(b * HW + y) * HW + x) * COUT + c;
        const float m  = m_buf[idx];
        const float va = var_buf[idx];
        const float s  = sqrtf(va + 1e-12f);
        const float uh = m * asc + ash;
        const float sh = s * ssc;
        const float z  = uh / (sh * 1.41421356237309515f + 1e-12f);
        const float u  = 0.5f * (1.0f + erff(z));
        su += u;
        sq += fmaxf(u * (1.0f - u), 1e-12f);
    }
    const size_t o = ((size_t)(b * COUT + c) * OUTHW + py) * OUTHW + px;
    outp[o]        = su * 0.25f;
    outp[o + OUT1] = sqrtf(sq * 0.25f) * 0.5f;
}

// ---------------------------------------------------------------------------
extern "C" void kernel_launch(void* const* d_in, const int* in_sizes, int n_in,
                              void* d_out, int out_size, void* d_ws, size_t ws_size,
                              hipStream_t stream)
{
    const float* mean   = (const float*)d_in[0];
    const float* stdv   = (const float*)d_in[1];
    const float* conv_w = (const float*)d_in[2];
    const float* conv_b = (const float*)d_in[3];
    const float* gamma  = (const float*)d_in[4];
    const float* beta   = (const float*)d_in[5];
    float* outp = (float*)d_out;

    const size_t NX = (size_t)BATCH * IMG * CIN;   // 12,845,056
    const size_t NW = (size_t)9 * COUT * CIN;      // 294,912
    const size_t NM = (size_t)MTOT * COUT;         // 25,690,112

    char* ws = (char*)d_ws;
    size_t off = 0;
    auto carve = [&](size_t bytes) -> char* {
        char* p = ws + off;
        off += (bytes + 255) & ~(size_t)255;
        return p;
    };
    _Float16* Xm      = (_Float16*)carve((NX + CIN) * 2);  // + zero row
    _Float16* Xv      = (_Float16*)carve((NX + CIN) * 2);  // + zero row
    _Float16* Wm      = (_Float16*)carve(NW * 2);
    _Float16* Wv      = (_Float16*)carve(NW * 2);
    float*    m_buf   = (float*)carve(NM * 4);
    float*    var_buf = (float*)carve(NM * 4);
    float*    parts   = (float*)carve((size_t)392 * 512 * 4);
    float*    a_scale = (float*)carve(256 * 4);
    float*    a_shift = (float*)carve(256 * 4);
    float*    s_scale = (float*)carve(256 * 4);
    (void)ws_size; (void)in_sizes; (void)n_in; (void)out_size;

    pack_x<<<dim3(BATCH * 49), 256, 0, stream>>>(mean, stdv, Xm, Xv);
    pack_w<<<dim3((9 * COUT * CIN) / 256), 256, 0, stream>>>(conv_w, Wm, Wv, Xm, Xv);

    gemm_conv<<<dim3(MTOT / 128, COUT / 128), 256, 0, stream>>>(
        Xm, Wm, conv_b, m_buf, 1.0f, 1);
    gemm_conv<<<dim3(MTOT / 128, COUT / 128), 256, 0, stream>>>(
        Xv, Wv, nullptr, var_buf, 1.0f / 4096.0f, 0);

    reduce_stage1<<<dim3(MTOT / 256), 256, 0, stream>>>(m_buf, parts);
    reduce_stage2<<<dim3(1), 256, 0, stream>>>(parts, gamma, beta,
                                               a_scale, a_shift, s_scale);

    compose_pool<<<dim3(BATCH * OUTHW * OUTHW), 256, 0, stream>>>(
        m_buf, var_buf, a_scale, a_shift, s_scale, outp);
}
